// RNNBase_86741159510117
// MI455X (gfx1250) — compile-verified, matrix operand order checked by reading, not run on hardware
//
#include <hip/hip_runtime.h>
#include <hip/hip_bf16.h>

#define DH   512
#define TSEQ 32768
#define L1   64
#define C1   (TSEQ / L1)   // 512 level-1 chunks
#define G1   (C1 / 16)     // 32 chain groups
#define L2   32
#define C2   (C1 / L2)     // 16 level-2 chunks

typedef __attribute__((ext_vector_type(16))) __bf16 v16bf;
typedef __attribute__((ext_vector_type(8)))  __bf16 v8bf;
typedef __attribute__((ext_vector_type(8)))  float  v8f;
typedef __attribute__((ext_vector_type(2)))  float  v2f;

__device__ __forceinline__ unsigned short f2bf(float f) {
  unsigned u = __float_as_uint(f);
  u += 0x7FFFu + ((u >> 16) & 1u);
  return (unsigned short)(u >> 16);
}

// ---------------------------------------------------------------------------
// f32 -> bf16 row-major conversion (x)
// ---------------------------------------------------------------------------
__global__ void __launch_bounds__(256)
cvt_f32_to_bf16(const float* __restrict__ src, unsigned short* __restrict__ dst, int n) {
  int i = (blockIdx.x * 256 + threadIdx.x) * 4;
  if (i < n) {
    float4 v = *(const float4*)(src + i);
    ushort4 o;
    o.x = f2bf(v.x); o.y = f2bf(v.y); o.z = f2bf(v.z); o.w = f2bf(v.w);
    *(ushort4*)(dst + i) = o;
  }
}

// ---------------------------------------------------------------------------
// Pack W (KxN f32 row-major) into bf16 WMMA B-operand order:
//   Wpk[((nt*16 + kit)*32 + lane)*16 + j] = bf16(W[k][n])
//   n = nt*16 + (lane&15); k = kit*32 + (lane>>4)*8 + (j&7) + (j>=8 ? 16 : 0)
// ---------------------------------------------------------------------------
__global__ void __launch_bounds__(256)
pack_w_bf16(const float* __restrict__ W, unsigned short* __restrict__ Wpk) {
  int t = blockIdx.x * 256 + threadIdx.x;   // 0..16383
  int lane = t & 31;
  int kit  = (t >> 5) & 15;
  int nt   = t >> 9;
  int n  = nt * 16 + (lane & 15);
  int kb = kit * 32 + ((lane >> 4) << 3);
  unsigned short* o = Wpk + (size_t)t * 16;
#pragma unroll
  for (int j = 0; j < 16; ++j) {
    int k = kb + (j & 7) + ((j >> 3) << 4);
    o[j] = f2bf(W[(size_t)k * DH + n]);
  }
}

// ---------------------------------------------------------------------------
// Pack M (512x512 f32 row-major) into f32 16x16x4 WMMA B-operand order
// for the scan step H_new = H * M^T + U   (B[k][n] = M[n][k]):
//   Mpk[((nt*128 + kit)*32 + lane)*2 + j] = M[n][k],
//   n = nt*16 + (lane&15); k = kit*4 + (lane>>4)*2 + j
// ---------------------------------------------------------------------------
__global__ void __launch_bounds__(256)
pack_f32_op(const float* __restrict__ M, float* __restrict__ Mpk) {
  int t = blockIdx.x * 256 + threadIdx.x;   // 0..131071
  int lane = t & 31;
  int kit  = (t >> 5) & 127;
  int nt   = t >> 12;
  int n = nt * 16 + (lane & 15);
  int k = kit * 4 + ((lane >> 4) << 1);
  float* o = Mpk + (size_t)t * 2;
  o[0] = M[(size_t)n * DH + k];
  o[1] = M[(size_t)n * DH + k + 1];
}

// ---------------------------------------------------------------------------
// Shared bf16 WMMA K-loop body: acc[2][4] += Abf-tile @ Bpk-tile over K=512
// ---------------------------------------------------------------------------
__device__ __forceinline__ void
gemm_bf16_kloop(const unsigned short* __restrict__ Abf,
                const unsigned short* __restrict__ Bpk,
                v8f acc[2][4], int arow0, int kh8, int ntg0) {
  for (int kit = 0; kit < 16; ++kit) {
    const int k0 = kit * 32 + kh8;
    v16bf a[2];
#pragma unroll
    for (int mt = 0; mt < 2; ++mt) {
      const unsigned short* ap = Abf + (size_t)(arow0 + mt * 16) * DH + k0;
      v8bf lo = *(const v8bf*)ap;
      v8bf hi = *(const v8bf*)(ap + 16);
#pragma unroll
      for (int i = 0; i < 8; ++i) { a[mt][i] = lo[i]; a[mt][8 + i] = hi[i]; }
    }
#pragma unroll
    for (int nt = 0; nt < 4; ++nt) {
      const unsigned short* bp = Bpk + (((size_t)(ntg0 + nt) * 16 + kit) * 32 + 0) * 16;
      v16bf b = *(const v16bf*)(bp + (size_t)(threadIdx.x & 31) * 16);
#pragma unroll
      for (int mt = 0; mt < 2; ++mt)
        acc[mt][nt] = __builtin_amdgcn_wmma_f32_16x16x32_bf16(
            false, a[mt], false, b, (short)0, acc[mt][nt], false, false);
    }
  }
}

__device__ __forceinline__ void
gemm_store(float* __restrict__ C, v8f acc[2][4], int mw, int nw, int lane) {
#pragma unroll
  for (int mt = 0; mt < 2; ++mt)
#pragma unroll
    for (int nt = 0; nt < 4; ++nt)
#pragma unroll
      for (int r = 0; r < 8; ++r) {
        int m = mw + mt * 16 + r + ((lane >> 4) << 3);
        int n = nw + nt * 16 + (lane & 15);
        C[(size_t)m * DH + n] = acc[mt][nt][r];
      }
}

// ---------------------------------------------------------------------------
// bf16 GEMM: C (Mx512 f32) = Abf (Mx512 row-major bf16) @ Bpk
// grid.x = M/128, grid.y = 4; 256 threads (8 waves, each 32 rows x 64 cols)
// ---------------------------------------------------------------------------
__global__ void __launch_bounds__(256)
gemm_bf16(const unsigned short* __restrict__ Abf,
          const unsigned short* __restrict__ Bpk,
          float* __restrict__ C) {
  const int lane = threadIdx.x & 31;
  const int w    = threadIdx.x >> 5;
  const int mw = blockIdx.x * 128 + (w >> 1) * 32;
  const int nw = blockIdx.y * 128 + (w & 1) * 64;
  v8f acc[2][4] = {};
  gemm_bf16_kloop(Abf, Bpk, acc, mw + (lane & 15), (lane >> 4) << 3, nw >> 4);
  gemm_store(C, acc, mw, nw, lane);
}

// ---------------------------------------------------------------------------
// Fused dual GEMM: C = A0@B0pk + A1@B1pk  (K = 2*512, no C read-back)
// ---------------------------------------------------------------------------
__global__ void __launch_bounds__(256)
gemm_bf16_dual(const unsigned short* __restrict__ A0,
               const unsigned short* __restrict__ B0pk,
               const unsigned short* __restrict__ A1,
               const unsigned short* __restrict__ B1pk,
               float* __restrict__ C) {
  const int lane = threadIdx.x & 31;
  const int w    = threadIdx.x >> 5;
  const int mw = blockIdx.x * 128 + (w >> 1) * 32;
  const int nw = blockIdx.y * 128 + (w & 1) * 64;
  v8f acc[2][4] = {};
  gemm_bf16_kloop(A0, B0pk, acc, mw + (lane & 15), (lane >> 4) << 3, nw >> 4);
  gemm_bf16_kloop(A1, B1pk, acc, mw + (lane & 15), (lane >> 4) << 3, nw >> 4);
  gemm_store(C, acc, mw, nw, lane);
}

// ---------------------------------------------------------------------------
// f32 512x512x512 matmul (repeated squaring of A): C = P @ Q
// grid 4x4, 256 threads; uses v_wmma_f32_16x16x4_f32
// ---------------------------------------------------------------------------
__global__ void __launch_bounds__(256)
matmul_f32(const float* __restrict__ P, const float* __restrict__ Q,
           float* __restrict__ C) {
  const int lane = threadIdx.x & 31;
  const int w    = threadIdx.x >> 5;
  const int mw = blockIdx.x * 128 + (w >> 1) * 32;
  const int nw = blockIdx.y * 128 + (w & 1) * 64;

  v8f acc[2][4] = {};
  const int ar = lane & 15;
  const int kh = (lane >> 4) << 1;

  for (int kit = 0; kit < 128; ++kit) {
    const int kk = kit * 4 + kh;
    v2f a[2];
#pragma unroll
    for (int mt = 0; mt < 2; ++mt)
      a[mt] = *(const v2f*)(P + (size_t)(mw + mt * 16 + ar) * DH + kk);
#pragma unroll
    for (int nt = 0; nt < 4; ++nt) {
      const int n = nw + nt * 16 + (lane & 15);
      v2f b;
      b.x = Q[(size_t)kk * DH + n];
      b.y = Q[(size_t)(kk + 1) * DH + n];
#pragma unroll
      for (int mt = 0; mt < 2; ++mt)
        acc[mt][nt] = __builtin_amdgcn_wmma_f32_16x16x4_f32(
            false, a[mt], false, b, (short)0, acc[mt][nt], false, false);
    }
  }
  gemm_store(C, acc, mw, nw, lane);
}

// ---------------------------------------------------------------------------
// Batched scan chains: each block owns 16 chains (M dim); state H in LDS.
// Step: H <- H @ M^T + U_row   (16x512x512 f32 WMMA GEMM per step)
//   U row for (chain, k) = U[(chain*L + k)*512 + n]
// seeds    (opt): initial states per chain (chain*512 + n)
// carry_out(opt): final states per chain
// pre_out  (opt): state BEFORE step k  -> rows (chain*L + k)  (level-2 pass C)
// post_bf  (opt): state AFTER step k as bf16 (level-1 pass C -> hs)
// ---------------------------------------------------------------------------
__global__ void __launch_bounds__(256)
scan_chain(const float* __restrict__ Mpk,
           const float* __restrict__ U,
           const float* __restrict__ seeds,
           float* __restrict__ carry_out,
           float* __restrict__ pre_out,
           unsigned short* __restrict__ post_bf,
           int L) {
  __shared__ float HL[16 * DH];
  const int tid  = threadIdx.x;
  const int lane = tid & 31;
  const int w    = tid >> 5;
  const int chainBase = blockIdx.x * 16;
  const int wn   = w * 64;            // wave column base
  const int ntg0 = wn >> 4;           // first global n-tile for this wave

  for (int e = tid; e < 16 * DH; e += 256) {
    int m = e >> 9, n = e & (DH - 1);
    HL[e] = seeds ? seeds[(size_t)(chainBase + m) * DH + n] : 0.0f;
  }
  __syncthreads();

  const int mrow = lane & 15;
  const int kh   = (lane >> 4) << 1;
  const float* wMpk = Mpk + (size_t)ntg0 * (128 * 32 * 2);

  for (int k = 0; k < L; ++k) {
    if (pre_out) {
      for (int e = tid; e < 16 * DH; e += 256) {
        int m = e >> 9, n = e & (DH - 1);
        pre_out[((size_t)(chainBase + m) * L + k) * DH + n] = HL[e];
      }
    }
    v8f acc[4] = {};
    for (int kit = 0; kit < 128; ++kit) {
      const int kk = (kit << 2) + kh;
      v2f a;
      a.x = HL[mrow * DH + kk];
      a.y = HL[mrow * DH + kk + 1];
      const float* bp = wMpk + ((size_t)kit * 32 + lane) * 2;
#pragma unroll
      for (int t = 0; t < 4; ++t) {
        v2f b = *(const v2f*)(bp + (size_t)t * (128 * 32 * 2));
        acc[t] = __builtin_amdgcn_wmma_f32_16x16x4_f32(
            false, a, false, b, (short)0, acc[t], false, false);
      }
    }
    __syncthreads();   // all waves done reading old H
    // v = acc + U; update LDS state (no data-dependent branches inside)
#pragma unroll
    for (int t = 0; t < 4; ++t)
#pragma unroll
      for (int r = 0; r < 8; ++r) {
        int m = r + ((lane >> 4) << 3);
        int n = wn + t * 16 + (lane & 15);
        size_t row = (size_t)(chainBase + m) * L + k;
        float v = acc[t][r] + U[row * DH + n];
        acc[t][r] = v;                 // keep for optional bf16 store
        HL[m * DH + n] = v;
      }
    if (post_bf) {                     // uniform branch wrapping whole loop
#pragma unroll
      for (int t = 0; t < 4; ++t)
#pragma unroll
        for (int r = 0; r < 8; ++r) {
          int m = r + ((lane >> 4) << 3);
          int n = wn + t * 16 + (lane & 15);
          size_t row = (size_t)(chainBase + m) * L + k;
          post_bf[row * DH + n] = f2bf(acc[t][r]);
        }
    }
    __syncthreads();
  }

  if (carry_out) {
    for (int e = tid; e < 16 * DH; e += 256) {
      int m = e >> 9, n = e & (DH - 1);
      carry_out[(size_t)(chainBase + m) * DH + n] = HL[e];
    }
  }
}

// ---------------------------------------------------------------------------
// Level-3: single-chain scan of C2=16 states with M3 = A^2048 (plain matvec)
// seeds2[j] = state BEFORE consuming carry2[j]
// ---------------------------------------------------------------------------
__global__ void __launch_bounds__(256)
matvec_chain(const float* __restrict__ M3, const float* __restrict__ carry2,
             float* __restrict__ seeds2) {
  __shared__ float h[DH];
  const int tid = threadIdx.x;
  const int r0 = tid * 2;
  h[r0] = 0.0f; h[r0 + 1] = 0.0f;
  __syncthreads();
  for (int j = 0; j < C2; ++j) {
    seeds2[(size_t)j * DH + r0]     = h[r0];
    seeds2[(size_t)j * DH + r0 + 1] = h[r0 + 1];
    float a0 = 0.0f, a1 = 0.0f;
    const float* row = M3 + (size_t)r0 * DH;
    for (int k = 0; k < DH; ++k) {
      float hv = h[k];
      a0 += row[k] * hv;
      a1 += row[DH + k] * hv;
    }
    a0 += carry2[(size_t)j * DH + r0];
    a1 += carry2[(size_t)j * DH + r0 + 1];
    __syncthreads();
    h[r0] = a0; h[r0 + 1] = a1;
    __syncthreads();
  }
}

// ---------------------------------------------------------------------------
extern "C" void kernel_launch(void* const* d_in, const int* in_sizes, int n_in,
                              void* d_out, int out_size, void* d_ws, size_t ws_size,
                              hipStream_t stream) {
  (void)in_sizes; (void)n_in; (void)out_size; (void)ws_size;
  const float* x  = (const float*)d_in[0];
  const float* A  = (const float*)d_in[1];
  const float* WB = (const float*)d_in[2];
  const float* WC = (const float*)d_in[3];
  const float* WD = (const float*)d_in[4];
  float* out = (float*)d_out;

  char* p = (char*)d_ws;
  auto alloc = [&](size_t bytes) -> char* {
    char* r = p; p += (bytes + 255) & ~(size_t)255; return r;
  };
  unsigned short* xbf  = (unsigned short*)alloc((size_t)TSEQ * DH * 2);
  unsigned short* hsbf = (unsigned short*)alloc((size_t)TSEQ * DH * 2);
  float* inp  = (float*)alloc((size_t)TSEQ * DH * 4);
  unsigned short* wbpk = (unsigned short*)alloc((size_t)DH * DH * 2);
  unsigned short* wcpk = (unsigned short*)alloc((size_t)DH * DH * 2);
  unsigned short* wdpk = (unsigned short*)alloc((size_t)DH * DH * 2);
  float* apk   = (float*)alloc((size_t)DH * DH * 4);
  float* m2    = (float*)alloc((size_t)DH * DH * 4);
  float* m2pk  = (float*)alloc((size_t)DH * DH * 4);
  float* m3    = (float*)alloc((size_t)DH * DH * 4);
  float* pt1   = (float*)alloc((size_t)DH * DH * 4);
  float* pt2   = (float*)alloc((size_t)DH * DH * 4);
  float* carry1 = (float*)alloc((size_t)C1 * DH * 4);
  float* h1     = (float*)alloc((size_t)C1 * DH * 4);
  float* carry2 = (float*)alloc((size_t)C2 * DH * 4);
  float* seeds2 = (float*)alloc((size_t)C2 * DH * 4);

  const dim3 blk(256);
  // --- conversions / packing ---
  cvt_f32_to_bf16<<<(TSEQ * DH) / 1024, blk, 0, stream>>>(x, xbf, TSEQ * DH);
  pack_w_bf16<<<64, blk, 0, stream>>>(WB, wbpk);
  pack_w_bf16<<<64, blk, 0, stream>>>(WC, wcpk);
  pack_w_bf16<<<64, blk, 0, stream>>>(WD, wdpk);
  pack_f32_op<<<512, blk, 0, stream>>>(A, apk);

  // --- inp = x @ WB (bf16 WMMA, f32 accum) ---
  gemm_bf16<<<dim3(TSEQ / 128, DH / 128), blk, 0, stream>>>(xbf, wbpk, inp);

  // --- powers by repeated squaring: m2 = A^64, m3 = A^2048 ---
  const dim3 pg(4, 4);
  matmul_f32<<<pg, blk, 0, stream>>>(A,   A,   pt1);  // A^2
  matmul_f32<<<pg, blk, 0, stream>>>(pt1, pt1, pt2);  // A^4
  matmul_f32<<<pg, blk, 0, stream>>>(pt2, pt2, pt1);  // A^8
  matmul_f32<<<pg, blk, 0, stream>>>(pt1, pt1, pt2);  // A^16
  matmul_f32<<<pg, blk, 0, stream>>>(pt2, pt2, pt1);  // A^32
  matmul_f32<<<pg, blk, 0, stream>>>(pt1, pt1, m2);   // A^64
  pack_f32_op<<<512, blk, 0, stream>>>(m2, m2pk);
  matmul_f32<<<pg, blk, 0, stream>>>(m2,  m2,  pt1);  // A^128
  matmul_f32<<<pg, blk, 0, stream>>>(pt1, pt1, pt2);  // A^256
  matmul_f32<<<pg, blk, 0, stream>>>(pt2, pt2, pt1);  // A^512
  matmul_f32<<<pg, blk, 0, stream>>>(pt1, pt1, pt2);  // A^1024
  matmul_f32<<<pg, blk, 0, stream>>>(pt2, pt2, m3);   // A^2048

  // --- 3-level chunked scan ---
  // L1 pass A: zero-seeded local scans -> carry1
  scan_chain<<<G1, blk, 0, stream>>>(apk, inp, nullptr, carry1, nullptr, nullptr, L1);
  // L2 pass A over carry1 with A^64 -> carry2
  scan_chain<<<1, blk, 0, stream>>>(m2pk, carry1, nullptr, carry2, nullptr, nullptr, L2);
  // L3: 16 sequential matvecs with A^2048 -> seeds2
  matvec_chain<<<1, blk, 0, stream>>>(m3, carry2, seeds2);
  // L2 pass C: seeded, emit pre-states = level-1 chunk seeds h1
  scan_chain<<<1, blk, 0, stream>>>(m2pk, carry1, seeds2, nullptr, h1, nullptr, L2);
  // L1 pass C: seeded, emit all h_t as bf16 -> hsbf
  scan_chain<<<G1, blk, 0, stream>>>(apk, inp, h1, nullptr, nullptr, hsbf, L1);

  // --- out = x @ WD + hs @ WC (fused, single pass over out) ---
  gemm_bf16_dual<<<dim3(TSEQ / 128, DH / 128), blk, 0, stream>>>(
      xbf, wdpk, hsbf, wcpk, out);
}
